// GambaMulti_57526791962978
// MI455X (gfx1250) — matrix operand order repaired and checked by reading, not compiled, not run on hardware
//
#include <hip/hip_runtime.h>
#include <hip/hip_bf16.h>

typedef __attribute__((ext_vector_type(16))) __bf16 v16bf;
typedef __attribute__((ext_vector_type(8)))  float  v8f;
typedef __attribute__((ext_vector_type(4)))  unsigned int u32x4;
typedef __attribute__((ext_vector_type(8)))  int  i32x8;
typedef __attribute__((ext_vector_type(4)))  int  i32x4;

#define NPG 512
#define Hh 128
#define D2 256
#define DI 512
#define DS 128
#define DTR 16

#if defined(__has_builtin)
#  if __has_builtin(__builtin_amdgcn_tensor_load_to_lds) && \
      __has_builtin(__builtin_amdgcn_s_wait_tensorcnt)
#    define HAVE_TDM 1
#  endif
#endif
#ifndef HAVE_TDM
#  define HAVE_TDM 0
#endif

static __device__ __forceinline__ float sigmoidf_(float x) {
    return 1.f / (1.f + __expf(-x));
}

static __device__ __forceinline__ v8f wmma_bf16(v16bf a, v16bf b, v8f c) {
    return __builtin_amdgcn_wmma_f32_16x16x32_bf16(false, a, false, b, (short)0, c,
                                                   false, false);
}

#if HAVE_TDM
// Issue a TDM DMA of a 32(row) x 64(col) f32 tile (row stride = N elements)
// from global `gptr` into LDS byte offset `ldsOff`.  Wave-level op, EXEC
// ignored, tracked by TENSORcnt.  D# packing per CDNA5 ISA 8.3/8.4.
static __device__ __forceinline__ void tdm_load_tile(const float* gptr,
                                                     unsigned ldsOff, int N) {
    unsigned long long ga = (unsigned long long)(uintptr_t)gptr;
    u32x4 g0;
    g0[0] = 1u;                                     // count=1, user descriptor
    g0[1] = ldsOff;                                 // lds_addr  (bits 63:32)
    g0[2] = (unsigned)(ga & 0xFFFFFFFFu);           // global_addr[31:0]
    g0[3] = (unsigned)((ga >> 32) & 0x01FFFFFFu)    // global_addr[56:32]
          | (2u << 30);                             // type=2 ("image")
    unsigned long long stride = (unsigned long long)N;  // dim0 stride (elems)
    i32x8 g1;
    g1[0] = (int)(2u << 16);                        // data_size=2 -> 4 bytes
    g1[1] = (int)((64u & 0xFFFFu) << 16);           // tensor_dim0[15:0]=64
    g1[2] = (int)(((64u >> 16) & 0xFFFFu) | ((32u & 0xFFFFu) << 16)); // dim0 hi | dim1 lo
    g1[3] = (int)(((32u >> 16) & 0xFFFFu) | (64u << 16));  // dim1 hi | tile_dim0=64
    g1[4] = (int)(32u);                             // tile_dim1=32, tile_dim2=0
    g1[5] = (int)(stride & 0xFFFFFFFFu);            // tensor_dim0_stride[31:0]
    g1[6] = (int)((stride >> 32) & 0xFFFFu);        // stride[47:32] | d1stride lo
    g1[7] = 0;
    i32x4 z4 = {};
#if __clang_major__ >= 23
    i32x8 z8 = {};
    __builtin_amdgcn_tensor_load_to_lds(g0, g1, z4, z4, z8, 0);
#else
    __builtin_amdgcn_tensor_load_to_lds(g0, g1, z4, z4, 0);
#endif
}
#endif

// ---------------------------------------------------------------------------
// bf16 WMMA GEMM: C[M,N] = act(A[M,K](lda) @ B[K,N] + bias)
// Block = 128 threads (4 waves), tile = 64x64; wave w owns a 16x64 strip
// (4 accumulators sharing one A fragment).  B panel is DMA'd by the TDM
// into an LDS staging buffer one K-chunk ahead (TENSORcnt pipeline), then
// converted/transposed to bf16 [col][k] so each fragment is one 32-byte
// contiguous ds read.  Requires M % 64 == 0.  act: 0=none, 3=softplus
// ---------------------------------------------------------------------------
__global__ void gemm_bf16_wmma(const float* __restrict__ A, int lda,
                               const float* __restrict__ B,
                               const float* __restrict__ bias,
                               float* __restrict__ C,
                               int M, int N, int K, int act) {
    __shared__ __bf16 sBt[64 * 32];          // [col][k] bf16, 4 KB
#if HAVE_TDM
    __shared__ float  sStage[32 * 64];       // raw f32 staging for TDM, 8 KB
#endif
    unsigned* sBt32 = (unsigned*)sBt;
    const int lane = threadIdx.x & 31;
    const int wave = threadIdx.x >> 5;
    const int r    = lane & 15;
    const int half = lane >> 4;
    const int col0 = blockIdx.x * 64;
    const int row0 = blockIdx.y * 64 + wave * 16;
    const int row  = row0 + r;               // always < M (M % 64 == 0)
    const float* __restrict__ Arow = A + (size_t)row * lda;
    const bool ntail = (col0 + 64 > N);

    v8f acc[4] = {};

#if HAVE_TDM
    const unsigned stageOff = (unsigned)(uintptr_t)sStage;
    if (!ntail && 32 <= K && wave == 0)
        tdm_load_tile(B + (size_t)0 * N + col0, stageOff, N);
#endif

    for (int k0 = 0; k0 < K; k0 += 32) {
        const bool fast = (k0 + 32 <= K) && !ntail;
#if HAVE_TDM
        if (fast && wave == 0) __builtin_amdgcn_s_wait_tensorcnt((short)0);
#endif
        __syncthreads();                     // prev sBt reads done / TDM done
        if (fast) {
#if HAVE_TDM
            // convert + transpose staged f32 panel -> bf16 [col][k]
            for (int t = threadIdx.x; t < 1024; t += 128) {
                int cc = t & 63, kp = t >> 6;         // kp = K-pair 0..15
                float f0 = sStage[(kp * 2 + 0) * 64 + cc];
                float f1 = sStage[(kp * 2 + 1) * 64 + cc];
                union { __bf16 h[2]; unsigned u; } p;
                p.h[0] = (__bf16)f0; p.h[1] = (__bf16)f1;
                sBt32[cc * 16 + kp] = p.u;
            }
#else
            // vectorized global loads, packed convert, transposed store
#pragma unroll
            for (int i = 0; i < 2; i++) {
                int li = threadIdx.x + i * 128;       // 256 (kp,c4) slots
                int kp = li >> 4, c4 = (li & 15) << 2;
                const float* bp0 = B + (size_t)(k0 + 2 * kp) * N + col0 + c4;
                float4 v0 = *(const float4*)bp0;
                float4 v1 = *(const float4*)(bp0 + N);
                union { __bf16 h[2]; unsigned u; } p;
                p.h[0] = (__bf16)v0.x; p.h[1] = (__bf16)v1.x; sBt32[(c4+0)*16+kp] = p.u;
                p.h[0] = (__bf16)v0.y; p.h[1] = (__bf16)v1.y; sBt32[(c4+1)*16+kp] = p.u;
                p.h[0] = (__bf16)v0.z; p.h[1] = (__bf16)v1.z; sBt32[(c4+2)*16+kp] = p.u;
                p.h[0] = (__bf16)v0.w; p.h[1] = (__bf16)v1.w; sBt32[(c4+3)*16+kp] = p.u;
            }
#endif
        } else {
            for (int t = threadIdx.x; t < 2048; t += 128) {
                int kk = t & 31, cc = t >> 5;
                int gk = k0 + kk, gn = col0 + cc;
                float v = (gk < K && gn < N) ? B[(size_t)gk * N + gn] : 0.f;
                sBt[cc * 32 + kk] = (__bf16)v;
            }
        }
        __syncthreads();                     // sBt ready; stage buffer free
#if HAVE_TDM
        if ((k0 + 64 <= K) && !ntail && wave == 0)   // prefetch next chunk
            tdm_load_tile(B + (size_t)(k0 + 32) * N + col0, stageOff, N);
#endif

        // A fragment: lanes 0-15 K {0..15}, lanes 16-31 K {16..31}
        v16bf a;
        if (k0 + 32 <= K) {
            if (k0 + 32 < K) __builtin_prefetch(Arow + k0 + 32, 0, 1);
            const float4* Ap = (const float4*)(Arow + k0);
            float4 a0 = Ap[half * 4 + 0];
            float4 a1 = Ap[half * 4 + 1];
            float4 a2 = Ap[half * 4 + 2];
            float4 a3 = Ap[half * 4 + 3];
            a[0]  = (__bf16)a0.x; a[1]  = (__bf16)a0.y;
            a[2]  = (__bf16)a0.z; a[3]  = (__bf16)a0.w;
            a[4]  = (__bf16)a1.x; a[5]  = (__bf16)a1.y;
            a[6]  = (__bf16)a1.z; a[7]  = (__bf16)a1.w;
            a[8]  = (__bf16)a2.x; a[9]  = (__bf16)a2.y;
            a[10] = (__bf16)a2.z; a[11] = (__bf16)a2.w;
            a[12] = (__bf16)a3.x; a[13] = (__bf16)a3.y;
            a[14] = (__bf16)a3.z; a[15] = (__bf16)a3.w;
        } else {
#pragma unroll
            for (int i = 0; i < 8; i++) {
                int k1 = k0 + half * 16 + i;
                int k2 = k0 + half * 16 + 8 + i;
                a[i]     = (__bf16)((k1 < K) ? Arow[k1] : 0.f);
                a[8 + i] = (__bf16)((k2 < K) ? Arow[k2] : 0.f);
            }
        }

        // 4 sub-tiles: fragment = 32 contiguous bytes of LDS (one b128 pair)
#pragma unroll
        for (int n = 0; n < 4; n++) {
            const v16bf bfrag =
                *(const v16bf*)(sBt + (n * 16 + r) * 32 + half * 16);
            acc[n] = wmma_bf16(a, bfrag, acc[n]);
        }
    }

#pragma unroll
    for (int n = 0; n < 4; n++) {
        int cn = col0 + n * 16 + r;
        if (cn < N) {
            float bv = bias ? bias[cn] : 0.f;
#pragma unroll
            for (int i = 0; i < 8; i++) {
                int cm = row0 + half * 8 + i;   // C layout: VGPR i -> M=i (+8 hi half)
                float v = acc[n][i] + bv;
                if (act == 3) v = (v > 20.f) ? v : log1pf(__expf(v));
                C[(size_t)cm * N + cn] = v;
            }
        }
    }
}

// ---------------------------------------------------------------------------
// Row LayerNorm (+optional ReLU).  grid = rows, block = 128
// ---------------------------------------------------------------------------
__global__ void ln_kernel(const float* __restrict__ X, const float* __restrict__ g,
                          const float* __restrict__ b, float* __restrict__ Y,
                          int D, int relu) {
    __shared__ float s1[128], s2[128];
    const size_t row = blockIdx.x;
    const float* xr = X + row * D;
    float a = 0.f, q = 0.f;
    for (int c = threadIdx.x; c < D; c += 128) { float v = xr[c]; a += v; q += v * v; }
    s1[threadIdx.x] = a; s2[threadIdx.x] = q;
    __syncthreads();
    for (int st = 64; st > 0; st >>= 1) {
        if ((int)threadIdx.x < st) {
            s1[threadIdx.x] += s1[threadIdx.x + st];
            s2[threadIdx.x] += s2[threadIdx.x + st];
        }
        __syncthreads();
    }
    float mu  = s1[0] / D;
    float var = s2[0] / D - mu * mu;
    float inv = rsqrtf(var + 1e-5f);
    for (int c = threadIdx.x; c < D; c += 128) {
        float v = (xr[c] - mu) * inv * g[c] + b[c];
        if (relu) v = fmaxf(v, 0.f);
        Y[row * D + c] = v;
    }
}

__global__ void zero_kernel(float* __restrict__ p, size_t n) {
    for (size_t i = blockIdx.x * (size_t)blockDim.x + threadIdx.x; i < n;
         i += (size_t)gridDim.x * blockDim.x) p[i] = 0.f;
}

// scatter-add of 128-wide features along edges: out[dst] += feat[src]
__global__ void edge_scatter(const float* __restrict__ feat, const int* __restrict__ src,
                             const int* __restrict__ dst, float* __restrict__ out, int E) {
    size_t tid = blockIdx.x * (size_t)blockDim.x + threadIdx.x; // E*32 threads
    if (tid >= (size_t)E * 32) return;
    int e  = (int)(tid >> 5);
    int c4 = (int)(tid & 31);
    const float4 v = *(const float4*)(feat + (size_t)src[e] * Hh + c4 * 4);
    float* o = out + (size_t)dst[e] * Hh + c4 * 4;
    atomicAdd(o + 0, v.x); atomicAdd(o + 1, v.y);
    atomicAdd(o + 2, v.z); atomicAdd(o + 3, v.w);
}

// GRU cell, in-place h update.  N*128 threads.
__global__ void gru_kernel(const float* __restrict__ gi, const float* __restrict__ gh,
                           float* __restrict__ h, int N) {
    size_t t = blockIdx.x * (size_t)blockDim.x + threadIdx.x;
    if (t >= (size_t)N * Hh) return;
    size_t n = t / Hh; int c = (int)(t % Hh);
    const float* gin = gi + n * 3 * Hh;
    const float* ghn = gh + n * 3 * Hh;
    float r  = sigmoidf_(gin[c] + ghn[c]);
    float z  = sigmoidf_(gin[Hh + c] + ghn[Hh + c]);
    float nn = tanhf(gin[2 * Hh + c] + r * ghn[2 * Hh + c]);
    float hv = h[t];
    h[t] = (1.f - z) * nn + z * hv;
}

// Y[n] = [A[n](Da) | B[n](Db)]
__global__ void concat2_kernel(const float* __restrict__ A, int Da,
                               const float* __restrict__ B, int Db,
                               float* __restrict__ Y, int N) {
    int D = Da + Db;
    size_t t = blockIdx.x * (size_t)blockDim.x + threadIdx.x;
    if (t >= (size_t)N * D) return;
    size_t n = t / D; int c = (int)(t % D);
    Y[t] = (c < Da) ? A[n * Da + c] : B[n * Db + (c - Da)];
}

// argmax over nodes of <x_dense[b,n,:], thetaW[:,k]>  (sigmoid is monotone)
__global__ void theta_argmax_kernel(const float* __restrict__ xd,
                                    const float* __restrict__ thW,
                                    int* __restrict__ idx) {
    __shared__ float sv[256]; __shared__ int si[256];
    __shared__ float sW[D2];
    int b = blockIdx.x >> 2, k = blockIdx.x & 3;
    for (int c = threadIdx.x; c < D2; c += 256) sW[c] = thW[c * 4 + k];
    __syncthreads();
    float best = -1e30f; int bi = 0;
    for (int n = threadIdx.x; n < NPG; n += 256) {
        const float* xr = xd + ((size_t)b * NPG + n) * D2;
        float dot = 0.f;
        for (int c = 0; c < D2; c++) dot += xr[c] * sW[c];
        if (dot > best) { best = dot; bi = n; }
    }
    sv[threadIdx.x] = best; si[threadIdx.x] = bi;
    __syncthreads();
    for (int st = 128; st > 0; st >>= 1) {
        if ((int)threadIdx.x < st) {
            float ov = sv[threadIdx.x + st]; int oi = si[threadIdx.x + st];
            if (ov > sv[threadIdx.x] ||
                (ov == sv[threadIdx.x] && oi < si[threadIdx.x])) {
                sv[threadIdx.x] = ov; si[threadIdx.x] = oi;
            }
        }
        __syncthreads();
    }
    if (threadIdx.x == 0) idx[b * 4 + k] = si[0];
}

// tokens[b,k,:] = x_dense[b, idx[b,k], :] / NPG     (64*4*256 threads)
__global__ void token_gather_kernel(const float* __restrict__ xd,
                                    const int* __restrict__ idx,
                                    float* __restrict__ tok) {
    int t = blockIdx.x * blockDim.x + threadIdx.x;
    if (t >= 64 * 4 * D2) return;
    int b = t / (4 * D2); int rem = t % (4 * D2);
    int k = rem / D2;     int c = rem % D2;
    tok[t] = xd[((size_t)b * NPG + idx[b * 4 + k]) * D2 + c] * (1.f / NPG);
}

// causal depthwise conv (DCONV=4) + SiLU. rows = b*4+l over [256][512].
__global__ void conv_silu_kernel(const float* __restrict__ xz,
                                 const float* __restrict__ cw,
                                 const float* __restrict__ cb,
                                 float* __restrict__ xs) {
    int t = blockIdx.x * blockDim.x + threadIdx.x;
    if (t >= 256 * DI) return;
    int row = t / DI, d = t % DI, l = row & 3;
    float acc = cb[d];
#pragma unroll
    for (int k = 0; k < 4; k++) {
        int ls = l + k - 3;
        if (ls >= 0) acc += xz[(size_t)(row - l + ls) * (2 * DI) + d] * cw[d * 4 + k];
    }
    xs[t] = acc * sigmoidf_(acc);
}

// selective scan: one block per (b,d), 128 threads = states
__global__ void scan_kernel(const float* __restrict__ dt, const float* __restrict__ Alog,
                            const float* __restrict__ xp, const float* __restrict__ xs,
                            const float* __restrict__ Dp, const float* __restrict__ xz,
                            float* __restrict__ y) {
    __shared__ float red[128];
    int b = blockIdx.x / DI, d = blockIdx.x % DI, s = threadIdx.x;
    float Av = -__expf(Alog[(size_t)d * DS + s]);
    float h = 0.f;
    for (int l = 0; l < 4; l++) {
        int row = b * 4 + l;
        float dtv = dt[(size_t)row * DI + d];
        float xv  = xs[(size_t)row * DI + d];
        float Bv  = xp[(size_t)row * (DTR + 2 * DS) + DTR + s];
        float Cv  = xp[(size_t)row * (DTR + 2 * DS) + DTR + DS + s];
        h = __expf(dtv * Av) * h + dtv * Bv * xv;
        red[s] = h * Cv;
        __syncthreads();
        for (int st = 64; st > 0; st >>= 1) {
            if (s < st) red[s] += red[s + st];
            __syncthreads();
        }
        if (s == 0) {
            float zv = xz[(size_t)row * (2 * DI) + DI + d];
            y[(size_t)row * DI + d] = (red[0] + xv * Dp[d]) * (zv * sigmoidf_(zv));
        }
        __syncthreads();
    }
}

// cat2[n] = [x_orig[n](128) | xm[batch(n), L-1, :](256)]
__global__ void merge_concat_kernel(const float* __restrict__ xo,
                                    const float* __restrict__ xm,
                                    float* __restrict__ Y, int N) {
    size_t t = blockIdx.x * (size_t)blockDim.x + threadIdx.x;
    if (t >= (size_t)N * 384) return;
    size_t n = t / 384; int c = (int)(t % 384);
    if (c < Hh) Y[t] = xo[n * Hh + c];
    else        Y[t] = xm[((n >> 9) * 4 + 3) * (size_t)D2 + (c - Hh)];
}

// global add pool: graphs are contiguous blocks of NPG nodes
__global__ void pool_kernel(const float* __restrict__ x, float* __restrict__ out) {
    int b = blockIdx.x, c = threadIdx.x;   // 64 blocks x 128 threads
    float acc = 0.f;
    for (int n = 0; n < NPG; n++)
        acc += x[((size_t)b * NPG + n) * Hh + c];
    out[(size_t)b * Hh + c] = acc;
}

__global__ void add_kernel(const float* __restrict__ a, const float* __restrict__ b,
                           float* __restrict__ o, size_t n) {
    size_t t = blockIdx.x * (size_t)blockDim.x + threadIdx.x;
    if (t < n) o[t] = a[t] + b[t];
}

// ---------------------------------------------------------------------------
static inline void gemm(const float* A, int lda, const float* B, const float* bias,
                        float* C, int M, int N, int K, int act, hipStream_t s) {
    dim3 g((N + 63) / 64, M / 64);           // M % 64 == 0 for every call site
    gemm_bf16_wmma<<<g, 128, 0, s>>>(A, lda, B, bias, C, M, N, K, act);
}

extern "C" void kernel_launch(void* const* d_in, const int* in_sizes, int n_in,
                              void* d_out, int out_size, void* d_ws, size_t ws_size,
                              hipStream_t stream) {
    const float* X_IN = (const float*)d_in[0];
    const int*   EI   = (const int*)d_in[1];
    const int E = in_sizes[1] / 2;
    const int N = in_sizes[0] / Hh;              // 32768 nodes
    const int* SRC = EI, * DST = EI + E;

    // params in pytree (sorted-key) order after x, edge_index, batch
    const float* decW1 = (const float*)d_in[3];  const float* decW2 = (const float*)d_in[4];
    const float* decb1 = (const float*)d_in[5];  const float* decb2 = (const float*)d_in[6];
    const float* decbe1= (const float*)d_in[7];  const float* decbe2= (const float*)d_in[8];
    const float* decg1 = (const float*)d_in[9];  const float* decg2 = (const float*)d_in[10];
    const float* encW1 = (const float*)d_in[11]; const float* encW2 = (const float*)d_in[12];
    const float* encb1 = (const float*)d_in[13]; const float* encb2 = (const float*)d_in[14];
    const float* encbe1= (const float*)d_in[15]; const float* encbe2= (const float*)d_in[16];
    const float* encg1 = (const float*)d_in[17]; const float* encg2 = (const float*)d_in[18];
    const float* ggcW  = (const float*)d_in[19];
    const float* Whh   = (const float*)d_in[20]; const float* Wih   = (const float*)d_in[21];
    const float* bhh   = (const float*)d_in[22]; const float* bih   = (const float*)d_in[23];
    const float* lnm_b = (const float*)d_in[52]; const float* lnm_g = (const float*)d_in[53];
    const float* oginW = (const float*)d_in[54]; const float* oginb = (const float*)d_in[55];

    // workspace bump allocator
    float* w = (float*)d_ws;
    size_t off = 0;
    auto alloc = [&](size_t n) { float* p = w + off; off += (n + 255) & ~(size_t)255; return p; };
    float* xbuf  = alloc((size_t)N * Hh);
    float* xorig = alloc((size_t)N * Hh);
    float* pe    = alloc((size_t)N * Hh);
    float* t0    = alloc((size_t)N * Hh);
    float* mbuf  = alloc((size_t)N * Hh);
    float* agg   = alloc((size_t)N * Hh);
    float* xa    = alloc((size_t)N * Hh);
    float* ench  = alloc((size_t)N * Hh);
    float* gi    = alloc((size_t)N * 3 * Hh);
    float* gh    = alloc((size_t)N * 3 * Hh);
    float* xc    = alloc((size_t)N * D2);
    float* cat2  = alloc((size_t)N * 3 * Hh);
    float* tok   = alloc(256 * D2);
    float* xz    = alloc(256 * 2 * DI);
    float* xs    = alloc(256 * DI);
    float* xproj = alloc(256 * (DTR + 2 * DS));
    float* dtb_  = alloc(256 * DI);
    float* ybuf  = alloc(256 * DI);
    float* xm    = alloc(256 * D2);
    float* pooled= alloc(64 * Hh);
    float* dech  = alloc(64 * Hh);
    float* dech2 = alloc(64 * 16);
    int*   idxb  = (int*)alloc(64 * 4);

    const int TB = 256;
    auto blocks = [&](size_t n) { return (unsigned)((n + TB - 1) / TB); };

    // ---- encoder MLP2 ----
    gemm(X_IN, Hh, encW1, encb1, ench, N, Hh, Hh, 0, stream);
    ln_kernel<<<N, 128, 0, stream>>>(ench, encg1, encbe1, ench, Hh, 1);
    gemm(ench, Hh, encW2, encb2, xbuf, N, Hh, Hh, 0, stream);
    ln_kernel<<<N, 128, 0, stream>>>(xbuf, encg2, encbe2, xbuf, Hh, 0);
    hipMemcpyAsync(xorig, xbuf, (size_t)N * Hh * 4, hipMemcpyDeviceToDevice, stream);
    hipMemcpyAsync(pe,    xbuf, (size_t)N * Hh * 4, hipMemcpyDeviceToDevice, stream);

    // ---- GatedGraphConv: 8 iterations of GEMM + scatter + GRU ----
    for (int l = 0; l < 8; l++) {
        gemm(pe, Hh, ggcW + (size_t)l * Hh * Hh, nullptr, t0, N, Hh, Hh, 0, stream);
        zero_kernel<<<2048, TB, 0, stream>>>(mbuf, (size_t)N * Hh);
        edge_scatter<<<blocks((size_t)E * 32), TB, 0, stream>>>(t0, SRC, DST, mbuf, E);
        gemm(mbuf, Hh, Wih, bih, gi, N, 3 * Hh, Hh, 0, stream);
        gemm(pe,   Hh, Whh, bhh, gh, N, 3 * Hh, Hh, 0, stream);
        gru_kernel<<<blocks((size_t)N * Hh), TB, 0, stream>>>(gi, gh, pe, N);
    }

    // ---- 2 Gamba layers ----
    for (int layer = 0; layer < 2; layer++) {
        int base = 24 + layer * 14;
        const float* Alog = (const float*)d_in[base + 0];
        const float* Dp   = (const float*)d_in[base + 1];
        const float* cw   = (const float*)d_in[base + 2];
        const float* cb   = (const float*)d_in[base + 3];
        const float* dtW  = (const float*)d_in[base + 4];
        const float* dtbias=(const float*)d_in[base + 5];
        const float* inW  = (const float*)d_in[base + 6];
        const float* mgW  = (const float*)d_in[base + 7];
        const float* mgb  = (const float*)d_in[base + 8];
        const float* mgbe = (const float*)d_in[base + 9];
        const float* mgg  = (const float*)d_in[base + 10];
        const float* outW = (const float*)d_in[base + 11];
        const float* thW  = (const float*)d_in[base + 12];
        const float* xW   = (const float*)d_in[base + 13];

        concat2_kernel<<<blocks((size_t)N * D2), TB, 0, stream>>>(xbuf, Hh, pe, Hh, xc, N);
        theta_argmax_kernel<<<256, 256, 0, stream>>>(xc, thW, idxb);
        token_gather_kernel<<<blocks(64 * 4 * D2), TB, 0, stream>>>(xc, idxb, tok);

        // Mamba block on [256, D2] tokens
        gemm(tok, D2, inW, nullptr, xz, 256, 2 * DI, D2, 0, stream);
        conv_silu_kernel<<<blocks(256 * DI), TB, 0, stream>>>(xz, cw, cb, xs);
        gemm(xs, DI, xW, nullptr, xproj, 256, DTR + 2 * DS, DI, 0, stream);
        gemm(xproj, DTR + 2 * DS, dtW, dtbias, dtb_, 256, DI, DTR, 3 /*softplus*/, stream);
        scan_kernel<<<64 * DI, 128, 0, stream>>>(dtb_, Alog, xproj, xs, Dp, xz, ybuf);
        gemm(ybuf, DI, outW, nullptr, xm, 256, D2, DI, 0, stream);
        ln_kernel<<<256, 128, 0, stream>>>(xm, lnm_g, lnm_b, xm, D2, 0);

        merge_concat_kernel<<<blocks((size_t)N * 384), TB, 0, stream>>>(xorig, xm, cat2, N);
        gemm(cat2, 3 * Hh, mgW, mgb, t0, N, Hh, 3 * Hh, 0, stream);
        ln_kernel<<<N, 128, 0, stream>>>(t0, mgg, mgbe, xbuf, Hh, 0);
    }

    // ---- output GIN + pool + decoder ----
    zero_kernel<<<2048, TB, 0, stream>>>(agg, (size_t)N * Hh);
    edge_scatter<<<blocks((size_t)E * 32), TB, 0, stream>>>(xbuf, SRC, DST, agg, E);
    add_kernel<<<blocks((size_t)N * Hh), TB, 0, stream>>>(xbuf, agg, xa, (size_t)N * Hh);
    gemm(xa, Hh, oginW, oginb, t0, N, Hh, Hh, 0, stream);
    pool_kernel<<<64, 128, 0, stream>>>(t0, pooled);

    gemm(pooled, Hh, decW1, decb1, dech, 64, Hh, Hh, 0, stream);
    ln_kernel<<<64, 128, 0, stream>>>(dech, decg1, decbe1, dech, Hh, 1);
    gemm(dech, Hh, decW2, decb2, dech2, 64, 10, Hh, 0, stream);
    ln_kernel<<<64, 128, 0, stream>>>(dech2, decg2, decbe2, (float*)d_out, 10, 0);
}
